// MinGRU_20916490731573
// MI455X (gfx1250) — matrix operand order repaired and checked by reading, not compiled, not run on hardware
//
#include <hip/hip_runtime.h>

typedef __attribute__((ext_vector_type(16))) _Float16 v16h;
typedef __attribute__((ext_vector_type(8)))  _Float16 v8h;
typedef __attribute__((ext_vector_type(8)))  float    v8f;

#define B_      8
#define S_      4096
#define D_      1024
#define TWO_D   2048
#define L_      3
#define NTOK    (B_ * S_)        // 32768 tokens
#define CHUNK   64
#define NCHUNK  (S_ / CHUNK)     // 64 chunks

// ---------------------------------------------------------------------------
// Kernel 0: convert W (f32, row-major [l][k][n]) into f16 swizzled WMMA-B
// fragment layout. For each 32(K)x16(N) block, half index = lane*16 + j with
//   lane = (k%32 < 16) ? n%16 : 16 + n%16 ;  j = k%16
// so each lane's v16h B fragment is one contiguous 32-byte read.
// ---------------------------------------------------------------------------
__global__ void mingru_pack_w(const float* __restrict__ W,
                              _Float16* __restrict__ Wp) {
  int idx = blockIdx.x * 256 + threadIdx.x;        // over L_*D_*TWO_D
  int n = idx % TWO_D;
  int k = (idx / TWO_D) % D_;
  int l = idx / (TWO_D * D_);
  float w = W[idx];
  int kb = k >> 5, nb = n >> 4;
  int kl = k & 31, nl = n & 15;
  int lane = (kl < 16) ? nl : (16 + nl);
  int j = kl & 15;
  size_t dst = (size_t)l * D_ * TWO_D +
               ((size_t)(kb * (TWO_D / 16) + nb)) * 512 + lane * 16 + j;
  Wp[dst] = (_Float16)w;
}

// ---------------------------------------------------------------------------
// Kernel 1: per-token LayerNorm stats (mu, rstd). One wave32 per token.
// ---------------------------------------------------------------------------
__global__ void mingru_ln_stats(const float* __restrict__ x,
                                float2* __restrict__ stats) {
  int wave = threadIdx.x >> 5;
  int lane = threadIdx.x & 31;
  int token = blockIdx.x * 8 + wave;
  const float* row = x + (size_t)token * D_;
  float s = 0.f, ss = 0.f;
#pragma unroll
  for (int j = 0; j < 8; ++j) {
    float4 v = *(const float4*)(row + j * 128 + lane * 4);
    s  += v.x + v.y + v.z + v.w;
    ss += v.x * v.x + v.y * v.y + v.z * v.z + v.w * v.w;
  }
#pragma unroll
  for (int off = 16; off; off >>= 1) {
    s  += __shfl_xor(s, off, 32);
    ss += __shfl_xor(ss, off, 32);
  }
  if (lane == 0) {
    float mu  = s / (float)D_;
    float var = ss / (float)D_ - mu * mu;
    stats[token] = make_float2(mu, rsqrtf(var + 1e-5f));
  }
}

// ---------------------------------------------------------------------------
// Kernel 2: fused LN + GEMM: gh[token, 0:2048] = f16( LN(inp[token]) @ W + b )
// Block: 256 threads = 8 wave32s. Block tile M=128, N=128.
// Waves arranged 4(M) x 2(N); wave tile M=32, N=64:
//   2 A fragments (4 x ds_load_b128) + 4 B fragments (8 x global_load_b128)
//   feed 8 v_wmma per K-step -- half the B-side fetch of a 16x128 tile.
// A staged through LDS with LN applied during f32->f16 conversion.
// B fragments streamed from L2-resident pre-swizzled Wp.
// ---------------------------------------------------------------------------
__launch_bounds__(256)
__global__ void mingru_gemm_ln(const float* __restrict__ inp,
                               const float2* __restrict__ stats,
                               const float* __restrict__ gamma,
                               const float* __restrict__ beta,
                               const _Float16* __restrict__ Wp,
                               const float* __restrict__ bias,
                               _Float16* __restrict__ gh) {
  __shared__ _Float16 lds[128 * 32];
  const int tid  = threadIdx.x;
  const int wave = tid >> 5, lane = tid & 31;
  const int wm = wave & 3;        // M-position of wave (0..3) -> rows 32*wm
  const int wn = wave >> 2;       // N-position of wave (0..1) -> cols 64*wn
  const int m0 = blockIdx.x * 128;
  const int n0 = blockIdx.y * 128;

  // staging role: each thread converts 16 halfs of one row
  const int srow  = tid >> 1;         // 0..127
  const int shalf = tid & 1;          // k_local base 0 or 16
  const int stoken = m0 + srow;
  const float2 st = stats[stoken];
  const float* srowp = inp + (size_t)stoken * D_ + shalf * 16;

  // A-fragment LDS addresses for this lane (WMMA 16-bit A layout)
  const int arow0  = wm * 32 + (lane & 15);       // first 16-row subtile
  const int kbaseA = (lane < 16) ? 0 : 16;

  v8f zero = {};
  v8f acc[2][4];
#pragma unroll
  for (int mi = 0; mi < 2; ++mi)
#pragma unroll
    for (int nt = 0; nt < 4; ++nt) acc[mi][nt] = zero;

  for (int k = 0; k < D_; k += 32) {
    const float* src  = srowp + k;
    const float* gsrc = gamma + k + shalf * 16;
    const float* bsrc = beta  + k + shalf * 16;
    __builtin_prefetch((const void*)(src + 32), 0, 0);   // next K-slice of A
    _Float16 h16[16];
#pragma unroll
    for (int j = 0; j < 4; ++j) {
      float4 v  = *(const float4*)(src  + j * 4);
      float4 gv = *(const float4*)(gsrc + j * 4);
      float4 bv = *(const float4*)(bsrc + j * 4);
      h16[j * 4 + 0] = (_Float16)((v.x - st.x) * st.y * gv.x + bv.x);
      h16[j * 4 + 1] = (_Float16)((v.y - st.x) * st.y * gv.y + bv.y);
      h16[j * 4 + 2] = (_Float16)((v.z - st.x) * st.y * gv.z + bv.z);
      h16[j * 4 + 3] = (_Float16)((v.w - st.x) * st.y * gv.w + bv.w);
    }
    __syncthreads();   // prev iteration's LDS reads complete
    *(v8h*)(&lds[srow * 32 + shalf * 16 + 0]) = *(v8h*)(&h16[0]);
    *(v8h*)(&lds[srow * 32 + shalf * 16 + 8]) = *(v8h*)(&h16[8]);
    __syncthreads();   // LDS tile visible

    v16h afrag[2];
#pragma unroll
    for (int mi = 0; mi < 2; ++mi) {
      const int arow = arow0 + mi * 16;
      v8h alo = *(const v8h*)(&lds[arow * 32 + kbaseA]);
      v8h ahi = *(const v8h*)(&lds[arow * 32 + kbaseA + 8]);
      afrag[mi] = __builtin_shufflevector(alo, ahi,
          0, 1, 2, 3, 4, 5, 6, 7, 8, 9, 10, 11, 12, 13, 14, 15);
    }

    const _Float16* wbase =
        Wp + ((size_t)(k >> 5) * (TWO_D / 16) + ((n0 + wn * 64) >> 4)) * 512 +
        lane * 16;
#pragma unroll
    for (int nt = 0; nt < 4; ++nt) {
      const _Float16* bp = wbase + nt * 512;
      v8h blo = *(const v8h*)(bp);
      v8h bhi = *(const v8h*)(bp + 8);
      v16h bfrag = __builtin_shufflevector(blo, bhi,
          0, 1, 2, 3, 4, 5, 6, 7, 8, 9, 10, 11, 12, 13, 14, 15);
#pragma unroll
      for (int mi = 0; mi < 2; ++mi) {
        acc[mi][nt] = __builtin_amdgcn_wmma_f32_16x16x32_f16(
            false, afrag[mi], false, bfrag, (short)0, acc[mi][nt], false, false);
      }
    }
  }

  // epilogue: +bias, store f16. C layout: VGPR r, lane<16 -> M=r, else M=8+r.
  const int ccol = lane & 15;
#pragma unroll
  for (int mi = 0; mi < 2; ++mi) {
    const int crow0 = wm * 32 + mi * 16 + ((lane < 16) ? 0 : 8);
#pragma unroll
    for (int nt = 0; nt < 4; ++nt) {
      const int col = n0 + wn * 64 + nt * 16 + ccol;
      const float bv = bias[col];
#pragma unroll
      for (int r = 0; r < 8; ++r) {
        const int token = m0 + crow0 + r;
        gh[(size_t)token * TWO_D + col] = (_Float16)(acc[mi][nt][r] + bv);
      }
    }
  }
}

// ---------------------------------------------------------------------------
// Scan helpers: h_t = a_t * h_{t-1} + v_t  with
//   a = sigmoid(-gate) = 1/(1+exp(gate)),  v = sigmoid(gate)*g(hidden)
// ---------------------------------------------------------------------------
__device__ __forceinline__ void mingru_av(float gate, float hid,
                                          float& a, float& v) {
  a = 1.f / (1.f + __expf(gate));
  float z = 1.f - a;
  float gt = (hid >= 0.f) ? (hid + 0.5f) : 1.f / (1.f + __expf(-hid));
  v = z * gt;
}

// Phase A: per-(b,d,chunk) local products/offsets
__global__ void mingru_scanA(const _Float16* __restrict__ gh,
                             float* __restrict__ cA,
                             float* __restrict__ cV) {
  int idx = blockIdx.x * 256 + threadIdx.x;  // b*NCHUNK*D + c*D + d
  int d = idx % D_;
  int c = (idx / D_) % NCHUNK;
  int b = idx / (D_ * NCHUNK);
  const _Float16* base = gh + ((size_t)(b * S_ + c * CHUNK)) * TWO_D;
  float A = 1.f, V = 0.f;
  for (int t = 0; t < CHUNK; ++t) {
    float a, v;
    mingru_av((float)base[(size_t)t * TWO_D + d],
              (float)base[(size_t)t * TWO_D + D_ + d], a, v);
    A *= a;
    V = a * V + v;
  }
  cA[idx] = A;
  cV[idx] = V;
}

// Phase B: sequential combine of 64 chunk carries per (b,d); h0 = g(0) = 0.5
__global__ void mingru_scanB(const float* __restrict__ cA,
                             const float* __restrict__ cV,
                             float* __restrict__ Hin) {
  int idx = blockIdx.x * 256 + threadIdx.x;  // b*D + d
  int d = idx % D_;
  int b = idx / D_;
  float H = 0.5f;
  for (int c = 0; c < NCHUNK; ++c) {
    size_t o = ((size_t)b * NCHUNK + c) * D_ + d;
    Hin[o] = H;
    H = cA[o] * H + cV[o];
  }
}

// Phase C: replay with true chunk-entry state; fused residual; last hidden
// (pre-residual) to d_out tail. Safe in-place when inp == out.
__global__ void mingru_scanC(const _Float16* __restrict__ gh,
                             const float* __restrict__ inp,
                             const float* __restrict__ Hin,
                             float* __restrict__ out,
                             float* __restrict__ hlast) {
  int idx = blockIdx.x * 256 + threadIdx.x;
  int d = idx % D_;
  int c = (idx / D_) % NCHUNK;
  int b = idx / (D_ * NCHUNK);
  float h = Hin[((size_t)b * NCHUNK + c) * D_ + d];
  const _Float16* base  = gh  + ((size_t)(b * S_ + c * CHUNK)) * TWO_D;
  const float*    ibase = inp + ((size_t)(b * S_ + c * CHUNK)) * D_;
  float*          obase = out + ((size_t)(b * S_ + c * CHUNK)) * D_;
  for (int t = 0; t < CHUNK; ++t) {
    float a, v;
    mingru_av((float)base[(size_t)t * TWO_D + d],
              (float)base[(size_t)t * TWO_D + D_ + d], a, v);
    h = a * h + v;
    obase[(size_t)t * D_ + d] = h + ibase[(size_t)t * D_ + d];
  }
  if (c == NCHUNK - 1) hlast[b * D_ + d] = h;
}

// ---------------------------------------------------------------------------
extern "C" void kernel_launch(void* const* d_in, const int* in_sizes, int n_in,
                              void* d_out, int out_size, void* d_ws, size_t ws_size,
                              hipStream_t stream) {
  const float* x     = (const float*)d_in[0];  // (B,S,D)
  const float* gamma = (const float*)d_in[1];  // (L,D)
  const float* beta  = (const float*)d_in[2];  // (L,D)
  const float* W     = (const float*)d_in[3];  // (L,D,2D)
  const float* bias  = (const float*)d_in[4];  // (L,2D)
  float* out = (float*)d_out;                  // out (B,S,D) ++ h1,h2,h3 (B,D)

  char* ws = (char*)d_ws;
  _Float16* Wp    = (_Float16*)ws; ws += (size_t)L_ * D_ * TWO_D * sizeof(_Float16);
  float2*   stats = (float2*)ws;   ws += (size_t)NTOK * sizeof(float2);
  float*    cA    = (float*)ws;    ws += (size_t)B_ * NCHUNK * D_ * sizeof(float);
  float*    cV    = (float*)ws;    ws += (size_t)B_ * NCHUNK * D_ * sizeof(float);
  float*    Hin   = (float*)ws;    ws += (size_t)B_ * NCHUNK * D_ * sizeof(float);
  float*    act   = (float*)ws;    ws += (size_t)NTOK * D_ * sizeof(float);
  _Float16* gh    = (_Float16*)ws; ws += (size_t)NTOK * TWO_D * sizeof(_Float16);

  mingru_pack_w<<<(L_ * D_ * TWO_D) / 256, 256, 0, stream>>>(W, Wp);

  const int nScan = B_ * NCHUNK * D_;   // 524288
  for (int l = 0; l < L_; ++l) {
    const float* inp  = (l == 0) ? x : act;
    float*       lout = (l == L_ - 1) ? out : act;

    mingru_ln_stats<<<NTOK / 8, 256, 0, stream>>>(inp, stats);

    dim3 g(NTOK / 128, TWO_D / 128);
    mingru_gemm_ln<<<g, 256, 0, stream>>>(inp, stats,
                                          gamma + (size_t)l * D_,
                                          beta  + (size_t)l * D_,
                                          Wp + (size_t)l * D_ * TWO_D,
                                          bias + (size_t)l * TWO_D, gh);

    mingru_scanA<<<nScan / 256, 256, 0, stream>>>(gh, cA, cV);
    mingru_scanB<<<(B_ * D_) / 256, 256, 0, stream>>>(cA, cV, Hin);
    mingru_scanC<<<nScan / 256, 256, 0, stream>>>(
        gh, inp, Hin, lout, out + (size_t)NTOK * D_ + (size_t)l * B_ * D_);
  }
}
